// HCA_76828374991641
// MI455X (gfx1250) — matrix operand order repaired and checked by reading, not compiled, not run on hardware
//
#include <hip/hip_runtime.h>

// ---------------------------------------------------------------- types
typedef __attribute__((ext_vector_type(8)))  float  v8f;
typedef __attribute__((ext_vector_type(8)))  __bf16 v8bf;
typedef __attribute__((ext_vector_type(16))) __bf16 v16bf;

union ABf {           // one WMMA 16-bit A/B fragment (16 bf16 per lane)
  v16bf  v;
  v8bf   h[2];
  __bf16 e[16];
};

// ---------------------------------------------------------------- consts
constexpr int BATCH = 2;
constexpr int TLEN  = 2048;
constexpr int HID   = 2048;
constexpr int QCR   = 512;
constexpr int NH    = 16;
constexpr int DH    = 128;
constexpr int MBLK  = 16;
constexpr int NBLK  = 128;     // TLEN / MBLK
constexpr int WINSZ = 512;
constexpr int GRP   = 4;
constexpr int INTERF= 2048;
constexpr int ROWS  = BATCH * TLEN;   // 4096

constexpr int CKST  = 136;     // ck^T LDS row stride (bank 4*lo, 16B aligned)
constexpr int VST   = 40;      // v^T  LDS row stride, 32 keys + pad (bank 20*lo)
constexpr int BST   = 40;      // GEMM B-tile LDS row stride (bank 20*lo)

__device__ __forceinline__ __bf16 f2bf(float f) {
  unsigned u = __builtin_bit_cast(unsigned, f);
  unsigned r = u + 0x7FFFu + ((u >> 16) & 1u);      // round-to-nearest-even
  unsigned short s = (unsigned short)(r >> 16);
  return __builtin_bit_cast(__bf16, s);
}
__device__ __forceinline__ void store_val(float* p, float v)  { *p = v; }
__device__ __forceinline__ void store_val(__bf16* p, float v) { *p = f2bf(v); }

// exchange with lane^16 (cross half) -- VALU permlane, not ds_bpermute
__device__ __forceinline__ float xhalf(float x) {
#if __has_builtin(__builtin_amdgcn_permlanex16)
  int i = __builtin_bit_cast(int, x);
  i = __builtin_amdgcn_permlanex16(i, i, 0x76543210, 0xFEDCBA98, false, false);
  return __builtin_bit_cast(float, i);
#else
  return __shfl_xor(x, 16, 32);
#endif
}
__device__ __forceinline__ float fast_tanh(float x) {
#if __has_builtin(__builtin_amdgcn_tanhf)
  return __builtin_amdgcn_tanhf(x);                 // v_tanh_f32
#else
  return tanhf(x);
#endif
}

// ---------------------------------------------------------------- converts
__global__ void k_cvt(const float* __restrict__ s, __bf16* __restrict__ d, int n) {
  int i = blockIdx.x * 256 + threadIdx.x;
  if (i < n) d[i] = f2bf(s[i]);
}
// dst[c*R + r] = src[r*C + c]   (f32 -> bf16 transpose, one-time)
__global__ void k_tconv(const float* __restrict__ s, __bf16* __restrict__ d, int R, int C) {
  int i = blockIdx.x * 256 + threadIdx.x;
  if (i < R * C) {
    int r = i / C, c = i - r * C;
    d[(size_t)c * R + r] = f2bf(s[i]);
  }
}

// ---------------------------------------------------------------- GEMM
// C[M,N] = A[M,K] @ Bt[N,K]^T ; A,Bt bf16 row-major. 128 threads = 4 waves,
// wave = 16 rows x 64 cols, block = 64x64. grid = (N/64, M/64).
// B tile (64x32) double-buffered in LDS, shared by the 4 waves.
template <typename OutT>
__global__ void __launch_bounds__(128) k_gemm(
    const __bf16* __restrict__ A, int lda,
    const __bf16* __restrict__ Bt, int ldb,
    OutT* __restrict__ C, int ldc, int K)
{
  __shared__ alignas(32) __bf16 Bs[2][64 * BST];
  const int tid = threadIdx.x;
  const int lane = tid & 31, w = tid >> 5;
  const int lo = lane & 15, hi = lane >> 4;
  const int row0 = blockIdx.y * 64 + w * 16;
  const int col0 = blockIdx.x * 64;
  const int srow = tid >> 1, shalf = (tid & 1) * 16;
  const __bf16* bsrc = Bt + (size_t)(col0 + srow) * ldb + shalf;
  __bf16* bdst = &Bs[0][srow * BST + shalf];

  v8f acc[4];
  #pragma unroll
  for (int j = 0; j < 4; ++j)
    #pragma unroll
    for (int r = 0; r < 8; ++r) acc[j][r] = 0.f;

  *(v8bf*)(bdst)     = *(const v8bf*)(bsrc);         // prologue: stage k0=0
  *(v8bf*)(bdst + 8) = *(const v8bf*)(bsrc + 8);

  const __bf16* arow = A + (size_t)(row0 + lo) * lda;
  int buf = 0;
  for (int k0 = 0; k0 < K; k0 += 32, buf ^= 1) {
    __syncthreads();                                 // staged buf visible
    if (k0 + 32 < K) {                               // stage next buffer
      const __bf16* s = bsrc + k0 + 32;
      __bf16* d = &Bs[buf ^ 1][srow * BST + shalf];
      *(v8bf*)(d)     = *(const v8bf*)(s);
      *(v8bf*)(d + 8) = *(const v8bf*)(s + 8);
    }
    ABf a;                                           // A 16x32 frag
    a.h[0] = *(const v8bf*)(arow + k0 + hi * 8);
    a.h[1] = *(const v8bf*)(arow + k0 + 16 + hi * 8);
    __builtin_prefetch(arow + k0 + 256, 0, 1);       // global_prefetch_b8
    ABf bq[4];                                       // all B frags first
    #pragma unroll
    for (int j = 0; j < 4; ++j) {
      const __bf16* brow = &Bs[buf][(16 * j + lo) * BST + hi * 8];
      bq[j].h[0] = *(const v8bf*)(brow);
      bq[j].h[1] = *(const v8bf*)(brow + 16);
    }
    #pragma unroll
    for (int j = 0; j < 4; ++j)                      // then WMMA burst
      acc[j] = __builtin_amdgcn_wmma_f32_16x16x32_bf16(
          false, a.v, false, bq[j].v, (short)0, acc[j], false, false);
  }
  #pragma unroll
  for (int j = 0; j < 4; ++j) {
    int col = col0 + 16 * j + lo;
    #pragma unroll
    for (int r = 0; r < 8; ++r)
      store_val(&C[(size_t)(row0 + r + 8 * hi) * ldc + col], acc[j][r]);
  }
}

// ---------------------------------------------------------------- rope / norm
__global__ void k_rope_q(const float* __restrict__ q, const float* __restrict__ w,
                         __bf16* __restrict__ qb)
{
  int gw   = (blockIdx.x * blockDim.x + threadIdx.x) >> 5;
  int lane = threadIdx.x & 31;
  int hh = gw & (NH - 1);
  int row = gw >> 4;                         // 0..4095
  int b = row >> 11, t = row & (TLEN - 1);
  const float* src = q + (size_t)row * (NH * DH) + hh * DH;
  float x0 = src[lane], x1 = src[lane + 32], x2 = src[lane + 64], x3 = src[lane + 96];
  float inv = __powf(10000.0f, -(float)lane * (1.0f / 32.0f));
  float ang = (float)t * inv;
  float cs = cosf(ang), sn = sinf(ang);
  float r0 = x0 * cs - x1 * sn;
  float r1 = x0 * sn + x1 * cs;
  float ss = r0 * r0 + r1 * r1 + x2 * x2 + x3 * x3;
  #pragma unroll
  for (int o = 16; o; o >>= 1) ss += __shfl_xor(ss, o, 32);
  float sc = rsqrtf(ss * (1.0f / DH) + 1e-6f);
  __bf16* dst = qb + (((size_t)(b * NH + hh) * TLEN + t) * DH);   // (b,H,t,D)
  dst[lane]      = f2bf(r0 * sc * w[lane]);
  dst[lane + 32] = f2bf(r1 * sc * w[lane + 32]);
  dst[lane + 64] = f2bf(x2 * sc * w[lane + 64]);
  dst[lane + 96] = f2bf(x3 * sc * w[lane + 96]);
}

__global__ void k_rope_k(const float* __restrict__ kraw, const float* __restrict__ w,
                         __bf16* __restrict__ kb)
{
  int gw   = (blockIdx.x * blockDim.x + threadIdx.x) >> 5;  // row 0..4095
  int lane = threadIdx.x & 31;
  int t = gw & (TLEN - 1);
  const float* src = kraw + (size_t)gw * DH;
  float x0 = src[lane], x1 = src[lane + 32], x2 = src[lane + 64], x3 = src[lane + 96];
  float inv = __powf(10000.0f, -(float)lane * (1.0f / 32.0f));
  float ang = (float)t * inv;
  float cs = cosf(ang), sn = sinf(ang);
  float r0 = x0 * cs - x1 * sn;
  float r1 = x0 * sn + x1 * cs;
  float ss = r0 * r0 + r1 * r1 + x2 * x2 + x3 * x3;
  #pragma unroll
  for (int o = 16; o; o >>= 1) ss += __shfl_xor(ss, o, 32);
  float sc = rsqrtf(ss * (1.0f / DH) + 1e-6f);
  __bf16* dst = kb + (size_t)gw * DH;
  dst[lane]      = f2bf(r0 * sc * w[lane]);
  dst[lane + 32] = f2bf(r1 * sc * w[lane + 32]);
  dst[lane + 64] = f2bf(x2 * sc * w[lane + 64]);
  dst[lane + 96] = f2bf(x3 * sc * w[lane + 96]);
}

__global__ void k_norm_ck(const float* __restrict__ ckraw, const float* __restrict__ w,
                          __bf16* __restrict__ ckb)
{
  int gw   = (blockIdx.x * blockDim.x + threadIdx.x) >> 5;  // 0..255
  int lane = threadIdx.x & 31;
  const float* src = ckraw + (size_t)gw * DH;
  float x0 = src[lane], x1 = src[lane + 32], x2 = src[lane + 64], x3 = src[lane + 96];
  float ss = x0 * x0 + x1 * x1 + x2 * x2 + x3 * x3;
  #pragma unroll
  for (int o = 16; o; o >>= 1) ss += __shfl_xor(ss, o, 32);
  float sc = rsqrtf(ss * (1.0f / DH) + 1e-6f);
  __bf16* dst = ckb + (size_t)gw * DH;
  dst[lane]      = f2bf(x0 * sc * w[lane]);
  dst[lane + 32] = f2bf(x1 * sc * w[lane + 32]);
  dst[lane + 64] = f2bf(x2 * sc * w[lane + 64]);
  dst[lane + 96] = f2bf(x3 * sc * w[lane + 96]);
}

// ---------------------------------------------------------------- attention
// Transposed orientation, 32 keys per softmax step:
//   S^T = K_tile x Q^T (two 16-key sub-tiles), out^T = V^T x P^T with K=32
//   fully populated (no zero padding in the PV WMMA).
// kr1 may be memory-clamped for the tail; masks always use true key ids.
template <bool COMP>
__device__ __forceinline__ void attn_tile32(
    const ABf qfr[4],
    const __bf16* __restrict__ kr0,   // keys key0   .. key0+15
    const __bf16* __restrict__ kr1,   // keys key0+16.. key0+31 (mem-clamped)
    const __bf16* vT, int vstride,    // 32-key transposed V in LDS
    int lo, int hi, int q0, int key0,
    float& m, float& l, v8f o[8])
{
  v8f s0, s1;
  #pragma unroll
  for (int r = 0; r < 8; ++r) { s0[r] = 0.f; s1[r] = 0.f; }
  {
    ABf kf[4];
    #pragma unroll
    for (int kk = 0; kk < 4; ++kk) {
      kf[kk].h[0] = *(const v8bf*)(kr0 + lo * DH + kk * 32 + hi * 16);
      kf[kk].h[1] = *(const v8bf*)(kr0 + lo * DH + kk * 32 + 8 + hi * 16);
    }
    #pragma unroll
    for (int kk = 0; kk < 4; ++kk)
      s0 = __builtin_amdgcn_wmma_f32_16x16x32_bf16(
          false, kf[kk].v, false, qfr[kk].v, (short)0, s0, false, false);
    #pragma unroll
    for (int kk = 0; kk < 4; ++kk) {
      kf[kk].h[0] = *(const v8bf*)(kr1 + lo * DH + kk * 32 + hi * 16);
      kf[kk].h[1] = *(const v8bf*)(kr1 + lo * DH + kk * 32 + 8 + hi * 16);
    }
    #pragma unroll
    for (int kk = 0; kk < 4; ++kk)
      s1 = __builtin_amdgcn_wmma_f32_16x16x32_bf16(
          false, kf[kk].v, false, qfr[kk].v, (short)0, s1, false, false);
  }
  const float scale = 0.08838834764831845f;           // 1/sqrt(128)
  const int t = q0 + lo;                              // this lane's query
  #pragma unroll
  for (int r = 0; r < 8; ++r) {
    float x0 = s0[r] * scale, x1 = s1[r] * scale;
    if (!COMP) {
      x0 = 50.0f * fast_tanh(x0 * 0.02f);             // logit cap, sw only
      x1 = 50.0f * fast_tanh(x1 * 0.02f);
    }
    int k0i = key0 + r + 8 * hi;
    int k1i = k0i + 16;
    bool v0 = COMP ? ((k0i * MBLK + MBLK - 1) < t)
                   : ((k0i <= t) && (t - k0i < WINSZ));
    bool v1 = COMP ? ((k1i * MBLK + MBLK - 1) < t)
                   : ((k1i <= t) && (t - k1i < WINSZ));
    s0[r] = v0 ? x0 : -1e30f;
    s1[r] = v1 ? x1 : -1e30f;
  }
  // per-query (column) softmax over 32 keys: 15 in-reg ops + 1 permlane
  float mx = fmaxf(s0[0], s1[0]);
  #pragma unroll
  for (int r = 1; r < 8; ++r) mx = fmaxf(mx, fmaxf(s0[r], s1[r]));
  mx = fmaxf(mx, xhalf(mx));
  float mnew = fmaxf(m, mx);
  float alpha = __expf(m - mnew);                     // -1e30-(-1e30)=0 -> 1
  float p0[8], p1[8], rs = 0.f;
  #pragma unroll
  for (int r = 0; r < 8; ++r) {
    p0[r] = (s0[r] <= -1e29f) ? 0.f : __expf(s0[r] - mnew);
    p1[r] = (s1[r] <= -1e29f) ? 0.f : __expf(s1[r] - mnew);
    rs += p0[r] + p1[r];
  }
  rs += xhalf(rs);
  l = l * alpha + rs;
  m = mnew;
  #pragma unroll
  for (int j = 0; j < 8; ++j)
    #pragma unroll
    for (int r = 0; r < 8; ++r) o[j][r] *= alpha;     // alpha uniform per lane

  // P^T B-frag, all 32 K slots real:
  // lanes<16: K0..15 (p0 own + partner); lanes>=16: K16..31 (p1 partner + own)
  ABf pf;
  #pragma unroll
  for (int r = 0; r < 8; ++r) {
    float p0x = xhalf(p0[r]);
    float p1x = xhalf(p1[r]);
    pf.e[r]     = f2bf(hi ? p1x   : p0[r]);
    pf.e[8 + r] = f2bf(hi ? p1[r] : p0x);
  }
  #pragma unroll
  for (int j = 0; j < 8; ++j) {                       // out^T d-tiles, K=32
    const __bf16* vp = vT + (size_t)(16 * j + lo) * vstride + hi * 16;
    ABf vf;
    vf.h[0] = *(const v8bf*)(vp);                     // keys 16hi..16hi+7
    vf.h[1] = *(const v8bf*)(vp + 8);                 // keys 16hi+8..16hi+15
    o[j] = __builtin_amdgcn_wmma_f32_16x16x32_bf16(
        false, vf.v, false, pf.v, (short)0, o[j], false, false);
  }
}

// compressed-block stream -> f32 partial (b,H,t,D). grid (32, NH, BATCH), 128 thr.
__global__ void __launch_bounds__(128, 1) k_attn_comp(
    const __bf16* __restrict__ qb,   // (b,H,t,D)
    const __bf16* __restrict__ ckb,  // (b,NB,D)
    const float* __restrict__ sink,
    float* __restrict__ part)        // (b,H,t,D)
{
  __shared__ alignas(32) __bf16 ckT[DH * CKST];       // ck^T (d, n), padded
  const int tid = threadIdx.x, lane = tid & 31, w = tid >> 5;
  const int lo = lane & 15, hi = lane >> 4;
  const int b = blockIdx.z, hh = blockIdx.y;
  const int qt = blockIdx.x * 4 + w, q0 = qt * 16;

  const __bf16* cks = ckb + (size_t)b * NBLK * DH;
  for (int i = tid; i < NBLK * DH; i += 128) {        // stage ck^T
    int n = i >> 7, d = i & 127;
    ckT[d * CKST + n] = cks[i];
  }
  __syncthreads();

  ABf qfr[4];                                         // Q rows as B-frags
  const __bf16* qrow = qb + ((size_t)(b * NH + hh) * TLEN + q0 + lo) * DH;
  #pragma unroll
  for (int kk = 0; kk < 4; ++kk)
    qfr[kk].v = *(const v16bf*)(qrow + kk * 32 + hi * 16);

  float m = -1e30f, l = 0.f;
  v8f o[8];
  #pragma unroll
  for (int j = 0; j < 8; ++j)
    #pragma unroll
    for (int r = 0; r < 8; ++r) o[j][r] = 0.f;

  for (int nb0 = 0; nb0 < qt; nb0 += 32)              // 32 blocks per step
    attn_tile32<true>(qfr, cks + (size_t)nb0 * DH, cks + (size_t)(nb0 + 16) * DH,
                      ckT + nb0, CKST, lo, hi, q0, nb0, m, l, o);

  const float sinkv = sink[hh];
  float mf = fmaxf(m, sinkv);
  float alpha = __expf(m - mf);                       // 0 if stream empty
  float denom = l * alpha + __expf(sinkv - mf);       // sink column
  float wgt = alpha / denom;
  float* prow = part + ((size_t)(b * NH + hh) * TLEN + q0 + lo) * DH;
  #pragma unroll
  for (int j = 0; j < 8; ++j)
    *(v8f*)(prow + 16 * j + 8 * hi) = o[j] * wgt;     // d = 16j+8hi+r
}

// sliding-window stream + add partial -> aob bf16 (b,t,H*D).
__global__ void __launch_bounds__(128, 1) k_attn_sw(
    const __bf16* __restrict__ qb,   // (b,H,t,D)
    const __bf16* __restrict__ kb,   // (b,t,D)
    const __bf16* __restrict__ vb,   // (b,t,D)
    const float* __restrict__ sink,
    const float* __restrict__ part,  // (b,H,t,D)
    __bf16* __restrict__ aob)        // (b,t,H*D)
{
  __shared__ alignas(32) __bf16 vTs[4][DH * VST];     // per-wave v^T, 32 keys
  const int tid = threadIdx.x, lane = tid & 31, w = tid >> 5;
  const int lo = lane & 15, hi = lane >> 4;
  const int b = blockIdx.z, hh = blockIdx.y;
  const int qt = blockIdx.x * 4 + w, q0 = qt * 16;

  ABf qfr[4];
  const __bf16* qrow = qb + ((size_t)(b * NH + hh) * TLEN + q0 + lo) * DH;
  #pragma unroll
  for (int kk = 0; kk < 4; ++kk)
    qfr[kk].v = *(const v16bf*)(qrow + kk * 32 + hi * 16);

  float m = -1e30f, l = 0.f;
  v8f o[8];
  #pragma unroll
  for (int j = 0; j < 8; ++j)
    #pragma unroll
    for (int r = 0; r < 8; ++r) o[j][r] = 0.f;

  const __bf16* krb = kb + (size_t)b * TLEN * DH;
  const __bf16* vrb = vb + (size_t)b * TLEN * DH;
  int lowkey = q0 - (WINSZ - 1); if (lowkey < 0) lowkey = 0;
  for (int kt = lowkey >> 4; kt <= qt; kt += 2) {
    int key0 = kt * 16;
    int key1 = key0 + 16;
    int key1r = (key1 <= TLEN - 16) ? key1 : (TLEN - 16);  // mem clamp (tail)
    for (int i = lane; i < 512; i += 32) {            // stage v^T (32 keys)
      int kr = i >> 4, ch = i & 15;
      int srcrow = (kr < 16) ? (key0 + kr) : (key1r + kr - 16);
      v8bf vv = *(const v8bf*)(vrb + (size_t)srcrow * DH + ch * 8);
      #pragma unroll
      for (int q = 0; q < 8; ++q) vTs[w][(ch * 8 + q) * VST + kr] = vv[q];
    }
    asm volatile("s_wait_dscnt 0" ::: "memory");      // per-wave staging done
    attn_tile32<false>(qfr, krb + (size_t)key0 * DH, krb + (size_t)key1r * DH,
                       vTs[w], VST, lo, hi, q0, key0, m, l, o);
  }
  const float sinkv = sink[hh];
  float mf = fmaxf(m, sinkv);
  float alpha = __expf(m - mf);
  float denom = l * alpha + __expf(sinkv - mf);
  float wgt = alpha / denom;
  const float* prow = part + ((size_t)(b * NH + hh) * TLEN + q0 + lo) * DH;
  __bf16* arow = aob + (size_t)(b * TLEN + q0 + lo) * (NH * DH) + hh * DH;
  #pragma unroll
  for (int j = 0; j < 8; ++j) {
    v8f v = o[j] * wgt + *(const v8f*)(prow + 16 * j + 8 * hi);
    v8bf vb16;
    #pragma unroll
    for (int r = 0; r < 8; ++r) vb16[r] = f2bf(v[r]);
    *(v8bf*)(arow + 16 * j + 8 * hi) = vb16;
  }
}

// ---------------------------------------------------------------- launch
extern "C" void kernel_launch(void* const* d_in, const int* in_sizes, int n_in,
                              void* d_out, int out_size, void* d_ws, size_t ws_size,
                              hipStream_t stream)
{
  (void)in_sizes; (void)n_in; (void)out_size; (void)ws_size;
  const float* h      = (const float*)d_in[0];
  const float* Wq_c   = (const float*)d_in[1];
  const float* Wq_up  = (const float*)d_in[2];
  const float* Wk     = (const float*)d_in[3];
  const float* Wv     = (const float*)d_in[4];
  const float* W_comp = (const float*)d_in[5];
  const float* q_norm = (const float*)d_in[6];
  const float* k_norm = (const float*)d_in[7];
  const float* sink   = (const float*)d_in[8];
  const float* Wg     = (const float*)d_in[9];
  const float* Wo     = (const float*)d_in[10];
  float* out = (float*)d_out;

  char* p = (char*)d_ws;
  auto alloc = [&](size_t bytes) -> void* {
    void* r = (void*)p; p += (bytes + 255) & ~(size_t)255; return r;
  };
  __bf16* hb     = (__bf16*)alloc((size_t)ROWS * HID * 2);
  __bf16* WqcT   = (__bf16*)alloc((size_t)QCR * HID * 2);
  __bf16* WqupT  = (__bf16*)alloc((size_t)(NH * DH) * QCR * 2);
  __bf16* WkT    = (__bf16*)alloc((size_t)DH * HID * 2);
  __bf16* WvT    = (__bf16*)alloc((size_t)DH * HID * 2);
  __bf16* WcompT = (__bf16*)alloc((size_t)DH * (MBLK * HID) * 2);
  __bf16* WgT    = (__bf16*)alloc((size_t)GRP * INTERF * (NH / GRP * DH) * 2);
  __bf16* WoT    = (__bf16*)alloc((size_t)HID * (GRP * INTERF) * 2);
  __bf16* qcb    = (__bf16*)alloc((size_t)ROWS * QCR * 2);
  float*  qf     = (float*) alloc((size_t)ROWS * NH * DH * 4);
  __bf16* qb     = (__bf16*)alloc((size_t)ROWS * NH * DH * 2);
  float*  kraw   = (float*) alloc((size_t)ROWS * DH * 4);
  __bf16* kbv    = (__bf16*)alloc((size_t)ROWS * DH * 2);
  __bf16* vbv    = (__bf16*)alloc((size_t)ROWS * DH * 2);
  float*  ckraw  = (float*) alloc((size_t)BATCH * NBLK * DH * 4);
  __bf16* ckb    = (__bf16*)alloc((size_t)BATCH * NBLK * DH * 2);
  float*  part   = (float*) alloc((size_t)ROWS * NH * DH * 4);
  __bf16* aob    = (__bf16*)alloc((size_t)ROWS * NH * DH * 2);
  __bf16* interb = (__bf16*)alloc((size_t)ROWS * GRP * INTERF * 2);

  // ---- one-time conversions (bf16 + transposed weights)
  k_cvt<<<(ROWS * HID + 255) / 256, 256, 0, stream>>>(h, hb, ROWS * HID);
  k_tconv<<<(HID * QCR + 255) / 256, 256, 0, stream>>>(Wq_c, WqcT, HID, QCR);
  k_tconv<<<(QCR * NH * DH + 255) / 256, 256, 0, stream>>>(Wq_up, WqupT, QCR, NH * DH);
  k_tconv<<<(HID * DH + 255) / 256, 256, 0, stream>>>(Wk, WkT, HID, DH);
  k_tconv<<<(HID * DH + 255) / 256, 256, 0, stream>>>(Wv, WvT, HID, DH);
  k_tconv<<<(MBLK * HID * DH + 255) / 256, 256, 0, stream>>>(W_comp, WcompT, MBLK * HID, DH);
  for (int g = 0; g < GRP; ++g)
    k_tconv<<<(512 * INTERF + 255) / 256, 256, 0, stream>>>(
        Wg + (size_t)g * 512 * INTERF, WgT + (size_t)g * INTERF * 512, 512, INTERF);
  k_tconv<<<(GRP * INTERF * HID + 255) / 256, 256, 0, stream>>>(Wo, WoT, GRP * INTERF, HID);

  // ---- projections (WMMA GEMMs)
  k_gemm<__bf16><<<dim3(QCR / 64, ROWS / 64), 128, 0, stream>>>(
      hb, HID, WqcT, HID, qcb, QCR, HID);                       // h @ Wq_c
  k_gemm<float><<<dim3(NH * DH / 64, ROWS / 64), 128, 0, stream>>>(
      qcb, QCR, WqupT, QCR, qf, NH * DH, QCR);                  // @ Wq_up
  k_gemm<float><<<dim3(DH / 64, ROWS / 64), 128, 0, stream>>>(
      hb, HID, WkT, HID, kraw, DH, HID);                        // h @ Wk
  k_gemm<__bf16><<<dim3(DH / 64, ROWS / 64), 128, 0, stream>>>(
      hb, HID, WvT, HID, vbv, DH, HID);                         // h @ Wv
  k_gemm<float><<<dim3(DH / 64, BATCH * NBLK / 64), 128, 0, stream>>>(
      hb, MBLK * HID, WcompT, MBLK * HID, ckraw, DH, MBLK * HID); // comp KV

  // ---- RoPE + RMSNorm
  k_rope_q<<<ROWS * NH / 8, 256, 0, stream>>>(qf, q_norm, qb);
  k_rope_k<<<ROWS / 8, 256, 0, stream>>>(kraw, k_norm, kbv);
  k_norm_ck<<<BATCH * NBLK / 8, 256, 0, stream>>>(ckraw, k_norm, ckb);

  // ---- dual-stream attention with sink (split to avoid VGPR spills)
  k_attn_comp<<<dim3(TLEN / MBLK / 4, NH, BATCH), 128, 0, stream>>>(
      qb, ckb, sink, part);
  k_attn_sw<<<dim3(TLEN / MBLK / 4, NH, BATCH), 128, 0, stream>>>(
      qb, kbv, vbv, sink, part, aob);

  // ---- grouped MLP + output projection
  for (int g = 0; g < GRP; ++g)
    k_gemm<__bf16><<<dim3(INTERF / 64, ROWS / 64), 128, 0, stream>>>(
        aob + g * 512, NH * DH, WgT + (size_t)g * INTERF * 512, 512,
        interb + g * INTERF, GRP * INTERF, 512);
  k_gemm<float><<<dim3(HID / 64, ROWS / 64), 128, 0, stream>>>(
      interb, GRP * INTERF, WoT, GRP * INTERF, out, HID, GRP * INTERF);
}
